// MA_30889404792899
// MI455X (gfx1250) — matrix operand order repaired and checked by reading, not compiled, not run on hardware
//
#include <hip/hip_runtime.h>

#define T_DIM 8192
#define DIN   512
#define DQK   64
#define DV    64

typedef __bf16        v16bf __attribute__((ext_vector_type(16)));
typedef float         v8f   __attribute__((ext_vector_type(8)));
typedef unsigned int  v4u   __attribute__((ext_vector_type(4)));
typedef float         v4f   __attribute__((ext_vector_type(4)));

union Frag {
    v4u q[2];               // two 16B chunks
    unsigned short us[16];  // raw bf16 bits
    v16bf v;                // WMMA operand view
};

// f32 -> bf16 bits, round-to-nearest-even
static __device__ __forceinline__ unsigned short f2bfu(float f) {
    unsigned u = __float_as_uint(f);
    u += 0x7FFFu + ((u >> 16) & 1u);
    return (unsigned short)(u >> 16);
}

// ---------------------------------------------------------------------------
// Kernel 0: pre-swizzle Wq/Wk/Wv (f32 row-major) into bf16 B-fragment layout:
//   wb[(((mat*4 + ni)*16 + kc)*32 + lane)*16 + e]
//   where K = kc*32 + (lane<16?0:16) + e,  N = ni*16 + (lane&15)
// One thread per element; 3*512*64 = 98304 elements (192 KiB bf16).
// ---------------------------------------------------------------------------
__global__ void __launch_bounds__(256)
prep_w_kernel(const float* __restrict__ Wq, const float* __restrict__ Wk,
              const float* __restrict__ Wv, unsigned short* __restrict__ wb)
{
    const int idx  = blockIdx.x * 256 + threadIdx.x;
    const int e    = idx & 15;
    const int lane = (idx >> 4) & 31;
    const int kc   = (idx >> 9) & 15;
    const int ni   = (idx >> 13) & 3;
    const int mat  = idx >> 15;
    const int K    = kc * 32 + ((lane < 16) ? 0 : 16) + e;
    const int N    = ni * 16 + (lane & 15);
    const float* W = (mat == 0) ? Wq : ((mat == 1) ? Wk : Wv);
    wb[idx] = f2bfu(W[(size_t)K * DQK + N]);
}

// ---------------------------------------------------------------------------
// Kernel 1: q = x@Wq, k = x@Wk, v = x@Wv   (bf16 out; v stored transposed)
// block = 128 threads (4 waves). Block owns 16 rows of t; wave w owns output
// column tile w for all three matrices. Weights come pre-swizzled from wb.
// ---------------------------------------------------------------------------
__global__ void __launch_bounds__(128)
proj_qkv_kernel(const float* __restrict__ x,
                const unsigned short* __restrict__ wb,
                unsigned short* __restrict__ qb,   // [T][64] bf16
                unsigned short* __restrict__ kb,   // [T][64] bf16
                unsigned short* __restrict__ vT)   // [64][T] bf16 (transposed)
{
    const int w  = threadIdx.x >> 5;   // wave id 0..3 = output col tile
    const int l  = threadIdx.x & 31;
    const int ln = l & 15;
    const int hi = (l < 16) ? 0 : 1;
    const int t0 = blockIdx.x * 16;

    const int arow = t0 + ln;
    const int ncol = w * 16 + ln;

    // per-wave fragment bases into the swizzled weight buffer
    const unsigned short* wq = wb + ((size_t)(0 * 4 + w) * 16) * 512 + (size_t)l * 16;
    const unsigned short* wk = wb + ((size_t)(1 * 4 + w) * 16) * 512 + (size_t)l * 16;
    const unsigned short* wv = wb + ((size_t)(2 * 4 + w) * 16) * 512 + (size_t)l * 16;

    v8f accq = {}; v8f acck = {}; v8f accv = {};

    for (int kc = 0; kc < DIN / 32; ++kc) {
        const int kbase = kc * 32;

        // A fragment from x (f32 -> bf16), 16x32 layout
        Frag a;
        {
            const int c0 = kbase + hi * 16;
            const float* xp = x + (size_t)arow * DIN + c0;
            const v4f f0 = *(const v4f*)(xp);
            const v4f f1 = *(const v4f*)(xp + 4);
            const v4f f2 = *(const v4f*)(xp + 8);
            const v4f f3 = *(const v4f*)(xp + 12);
#pragma unroll
            for (int e = 0; e < 4; ++e) {
                a.us[e]      = f2bfu(f0[e]);
                a.us[e + 4]  = f2bfu(f1[e]);
                a.us[e + 8]  = f2bfu(f2[e]);
                a.us[e + 12] = f2bfu(f3[e]);
            }
        }

        // B fragments: pre-swizzled, two aligned 16B loads each
        Frag bq, bk, bv;
        const size_t fo = (size_t)kc * (32 * 16);
        bq.q[0] = *(const v4u*)(wq + fo);  bq.q[1] = *(const v4u*)(wq + fo + 8);
        bk.q[0] = *(const v4u*)(wk + fo);  bk.q[1] = *(const v4u*)(wk + fo + 8);
        bv.q[0] = *(const v4u*)(wv + fo);  bv.q[1] = *(const v4u*)(wv + fo + 8);

        accq = __builtin_amdgcn_wmma_f32_16x16x32_bf16(false, a.v, false, bq.v,
                                                       (short)0, accq, false, false);
        acck = __builtin_amdgcn_wmma_f32_16x16x32_bf16(false, a.v, false, bk.v,
                                                       (short)0, acck, false, false);
        accv = __builtin_amdgcn_wmma_f32_16x16x32_bf16(false, a.v, false, bv.v,
                                                       (short)0, accv, false, false);
    }

    // q, k: row-major bf16 stores (C layout: m = r + hi*8, n = ln)
#pragma unroll
    for (int r = 0; r < 8; ++r) {
        const int t = t0 + r + hi * 8;
        qb[(size_t)t * DQK + ncol] = f2bfu(accq[r]);
        kb[(size_t)t * DQK + ncol] = f2bfu(acck[r]);
    }
    // v: transposed store -> 8 consecutive t per lane = one 16B store
    {
        v4u st;
#pragma unroll
        for (int r = 0; r < 4; ++r)
            st[r] = (unsigned)f2bfu(accv[2 * r]) |
                    ((unsigned)f2bfu(accv[2 * r + 1]) << 16);
        *(v4u*)(vT + (size_t)ncol * T_DIM + t0 + hi * 8) = st;
    }
}

// ---------------------------------------------------------------------------
// Kernel 2: out = ((q k^T) .* tril(l)) @ v   (flash-style, causal skip)
// block = 128 threads (4 waves), 32 t-rows per block, s-tiles of 64.
// Double-buffered LDS S-tile -> one barrier per iteration.
// Heavy blocks (large t0) scheduled first.
// ---------------------------------------------------------------------------
__global__ void __launch_bounds__(128)
attn_kernel(const unsigned short* __restrict__ qb,
            const unsigned short* __restrict__ kb,
            const unsigned short* __restrict__ vT,
            const float* __restrict__ lmat,
            float* __restrict__ out)
{
    __shared__ unsigned short Slds[2][32 * 72];   // double-buffered S tile

    const int w     = threadIdx.x >> 5;
    const int l     = threadIdx.x & 31;
    const int ln    = l & 15;
    const int hi    = (l < 16) ? 0 : 1;
    const int t0    = ((int)gridDim.x - 1 - (int)blockIdx.x) * 32;  // heavy first
    const int mi    = w >> 1;          // 0..1
    const int nbase = (w & 1) * 2;     // tiles nbase, nbase+1

    // Preload Q A-fragments for this wave's m-subtile (2 K-chunks of 32)
    Frag qf[2];
    {
        const unsigned short* qp = qb + (size_t)(t0 + mi * 16 + ln) * DQK;
#pragma unroll
        for (int kc = 0; kc < 2; ++kc) {
            const int c0 = kc * 32 + hi * 16;
            qf[kc].q[0] = *(const v4u*)(qp + c0);
            qf[kc].q[1] = *(const v4u*)(qp + c0 + 8);
        }
    }

    // Loop-invariant l row pointers (C-layout rows of this wave's m-subtile)
    const float* lrow[8];
#pragma unroll
    for (int r = 0; r < 8; ++r)
        lrow[r] = lmat + (size_t)(t0 + mi * 16 + r + hi * 8) * T_DIM;

    v8f oacc[2];
    oacc[0] = {}; oacc[1] = {};

    const int tmax = t0 + 31;
    for (int s0 = 0; s0 <= tmax; s0 += 64) {
        const int  bi       = (s0 >> 6) & 1;
        const bool notlast  = (s0 + 64 <= tmax);
        const bool unmasked = (s0 + 64 <= t0);     // uniform: tile fully below diag

        // ---- S = (Q K^T) .* tril(l), f32 accumulate, bf16 to LDS ----
#pragma unroll
        for (int j = 0; j < 2; ++j) {
            const int ni   = nbase + j;
            const int scol = s0 + ni * 16 + ln;     // B column = s index
            v8f acc = {};
#pragma unroll
            for (int kc = 0; kc < 2; ++kc) {
                Frag kf;   // B 32x16: lane holds 16 consecutive d at row scol
                const int c0 = kc * 32 + hi * 16;
                const unsigned short* kp = kb + (size_t)scol * DQK + c0;
                kf.q[0] = *(const v4u*)(kp);
                kf.q[1] = *(const v4u*)(kp + 8);
                acc = __builtin_amdgcn_wmma_f32_16x16x32_bf16(
                        false, qf[kc].v, false, kf.v, (short)0, acc, false, false);
            }
            if (unmasked) {            // fast path: no compares, no selects
#pragma unroll
                for (int r = 0; r < 8; ++r) {
                    const float lv = lrow[r][scol];
                    if (notlast) __builtin_prefetch(&lrow[r][scol + 64], 0, 1);
                    Slds[bi][(mi * 16 + r + hi * 8) * 72 + ni * 16 + ln] =
                        f2bfu(acc[r] * lv);
                }
            } else {                   // diagonal tile: tril mask (covers OOB too)
#pragma unroll
                for (int r = 0; r < 8; ++r) {
                    const int trow = t0 + mi * 16 + r + hi * 8;
                    float lv = 0.0f;
                    if (scol <= trow) lv = lrow[r][scol];
                    Slds[bi][(mi * 16 + r + hi * 8) * 72 + ni * 16 + ln] =
                        f2bfu(acc[r] * lv);
                }
            }
        }
        __syncthreads();   // single barrier: write -> read; parity protects reuse

        // ---- O += S_bf16 @ V ----
#pragma unroll
        for (int j = 0; j < 2; ++j) {
            const int ni2  = nbase + j;
            const int vcol = ni2 * 16 + ln;
#pragma unroll
            for (int kc = 0; kc < 2; ++kc) {
                Frag sf, vf;
                const int c0   = kc * 32 + hi * 16;    // A-layout: 16 consecutive s
                const int srow = mi * 16 + ln;
                sf.q[0] = *(const v4u*)&Slds[bi][srow * 72 + c0];
                sf.q[1] = *(const v4u*)&Slds[bi][srow * 72 + c0 + 8];
                const int sb = s0 + kc * 32 + hi * 16; // B-layout: 16 consecutive s
                const unsigned short* vp = vT + (size_t)vcol * T_DIM + sb;
                vf.q[0] = *(const v4u*)(vp);
                vf.q[1] = *(const v4u*)(vp + 8);
                oacc[j] = __builtin_amdgcn_wmma_f32_16x16x32_bf16(
                        false, sf.v, false, vf.v, (short)0, oacc[j], false, false);
            }
        }
    }

    // ---- store O (f32) ----
#pragma unroll
    for (int j = 0; j < 2; ++j) {
        const int col = (nbase + j) * 16 + ln;
#pragma unroll
        for (int r = 0; r < 8; ++r) {
            const int trow = t0 + mi * 16 + r + hi * 8;
            out[(size_t)trow * DV + col] = oacc[j][r];
        }
    }
}

// ---------------------------------------------------------------------------
extern "C" void kernel_launch(void* const* d_in, const int* in_sizes, int n_in,
                              void* d_out, int out_size, void* d_ws, size_t ws_size,
                              hipStream_t stream) {
    (void)in_sizes; (void)n_in; (void)out_size; (void)ws_size;
    const float* x  = (const float*)d_in[0];
    const float* Wq = (const float*)d_in[1];
    const float* Wk = (const float*)d_in[2];
    const float* Wv = (const float*)d_in[3];
    const float* lm = (const float*)d_in[4];
    float* out = (float*)d_out;

    // workspace: q (1 MiB) | k (1 MiB) | v^T (1 MiB) | swizzled W (192 KiB)
    unsigned short* qb = (unsigned short*)d_ws;
    unsigned short* kb = qb + (size_t)T_DIM * DQK;
    unsigned short* vT = kb + (size_t)T_DIM * DQK;
    unsigned short* wb = vT + (size_t)T_DIM * DV;

    prep_w_kernel<<<dim3(3 * DIN * DQK / 256), 256, 0, stream>>>(Wq, Wk, Wv, wb);
    proj_qkv_kernel<<<dim3(T_DIM / 16), 128, 0, stream>>>(x, wb, qb, kb, vT);
    attn_kernel<<<dim3(T_DIM / 32), 128, 0, stream>>>(qb, kb, vT, lm, out);
}
